// DecoderTopDown_52733608460809
// MI455X (gfx1250) — compile-verified, matrix operand order checked by reading, not run on hardware
//
#include <hip/hip_runtime.h>
#include <math.h>
#include <stdint.h>

// Problem constants (match reference)
#define Bc   128
#define Rc   36
#define Tc   40
#define Ec   1024
#define Vc   2048
#define H1c  1024
#define H2c  1024
#define PHc  256
#define VOCc 10000

// fused activation row widths
#define K1c  5120   // x1c = [h2(1024) | uv(2048) | emb(1024) | h1(1024)]
#define K2c  4096   // x2c = [av(2048) | h1n(1024) | h2(1024)]

typedef __attribute__((ext_vector_type(16))) __bf16 v16bf;
typedef __attribute__((ext_vector_type(8)))  __bf16 v8bf;
typedef __attribute__((ext_vector_type(8)))  float  v8f;

// ---------------------------------------------------------------------------
// CDNA5 async global->LDS copy (ASYNCcnt path, ISA 10.7 / 15.18.3 op 98).
// VDST = VGPR holding LDS byte address (addr[31:0] of the generic pointer),
// VADDR = 64-bit global address. Inline asm: portable across toolchains.
// ---------------------------------------------------------------------------
__device__ __forceinline__ void async_copy_b128(const __bf16* gsrc, __bf16* ldst) {
  unsigned lofs = (unsigned)(uintptr_t)(void*)ldst;   // LDS byte offset
  asm volatile("global_load_async_to_lds_b128 %0, %1, off"
               :: "v"(lofs), "v"(gsrc) : "memory");
}
__device__ __forceinline__ void wait_async0() {
  asm volatile("s_wait_asynccnt 0x0" ::: "memory");
}

// ---------------------------------------------------------------------------
// fp32 -> bf16 conversion (weights converted once per launch; ~96MB bf16 total
// -> fully L2-resident (192MB) across all 40 recurrent steps)
// ---------------------------------------------------------------------------
__global__ void k_f32_to_bf16(const float* __restrict__ src, __bf16* __restrict__ dst, int n) {
  int i = blockIdx.x * blockDim.x + threadIdx.x;
  if (i < n) dst[i] = (__bf16)src[i];
}

// pack two fp32 weight matrices [N,Ka],[N,Kb] into fused bf16 [N, Ka+Kb]
__global__ void k_pack_w(const float* __restrict__ wa_, const float* __restrict__ wb_,
                         __bf16* __restrict__ dst, int Ka, int Kb, long total) {
  long i = (long)blockIdx.x * blockDim.x + threadIdx.x;
  if (i >= total) return;
  int Kt = Ka + Kb;
  long n = i / Kt;
  int  j = (int)(i - n * Kt);
  float v = (j < Ka) ? wa_[n * Ka + j] : wb_[n * Kb + (j - Ka)];
  dst[i] = (__bf16)v;
}

// x1c[:, H2 : H2+V] = bf16(union_vfeats)  (step-invariant, filled once)
__global__ void k_x1_uv(__bf16* __restrict__ x1c, const float* __restrict__ uv) {
  int i = blockIdx.x * blockDim.x + threadIdx.x;
  if (i >= Bc * Vc) return;
  int b = i / Vc, j = i % Vc;
  x1c[(long)b * K1c + H2c + j] = (__bf16)uv[i];
}

// per step: x1c[:, H2+V : H2+V+E] = bf16(embedding row for caption[b,t])
__global__ void k_x1_emb(__bf16* __restrict__ x1c, const float* __restrict__ embedW,
                         const int* __restrict__ captions, int t) {
  int i = blockIdx.x * blockDim.x + threadIdx.x;
  if (i >= Bc * Ec) return;
  int b = i / Ec, j = i % Ec;
  int cap = captions[b * Tc + t];
  x1c[(long)b * K1c + H2c + Vc + j] = (__bf16)embedW[(long)cap * Ec + j];
}

// ---------------------------------------------------------------------------
// WMMA bf16 GEMM:  C[m,n] = sum_k A[m,k] * W[n,k]  (+bias[n]) (mask rows)
// Block = 8 waves sharing one 16-row A tile staged in LDS (double-buffered
// 16x64 chunks, padded row stride 72 halves -> conflict-free ds_read_b128).
// The NEXT chunk's A tile is fetched with GLOBAL_LOAD_ASYNC_TO_LDS_B128 at
// the TOP of each chunk, overlapping the whole compute phase; the buffer is
// published with s_wait_asynccnt 0 + barrier (ISA async protocol).
// Each wave computes a 16x64 strip (4 WMMA accumulators). Per chunk: all B
// global loads + A ds loads are hoisted, an asm memory-clobber pins them
// above the MMA block, then 8 WMMAs issue back-to-back.
// Operand layout per ISA: lane L holds row/col (L&15); K-halves selected by
// (L>>4): elems 0..7 -> K=8*(L>>4)+0..7, elems 8..15 -> +16.
// C layout: VGPR r -> row r + 8*(L>>4), col (L&15).
// EXEC is all-ones at every WMMA (no lane-divergent flow before the k-loop;
// partial N tiles handled by valid[] store guards only).
// ---------------------------------------------------------------------------
#define GEMM_NT 4
#define KCHUNK  64
#define KSTEPS  (KCHUNK / 32)   // 2
#define ASTRIDE 72              // 64 + 8 halves padding (16B)

__global__ __launch_bounds__(256)
void k_gemm_bf16(const __bf16* __restrict__ A, const __bf16* __restrict__ W,
                 const float* __restrict__ bias, float* __restrict__ C,
                 int M, int N, int K, long ldc,
                 const int* __restrict__ lengths, int t) {
  __shared__ __attribute__((aligned(16))) __bf16 As[2][16][ASTRIDE];

  const int tid  = threadIdx.x;
  const int lane = tid & 31;
  const int wid  = tid >> 5;
  const int ntiles = N >> 4;
  const int nbm    = (ntiles + 31) >> 5;          // column-blocks per mtile
  const int mtile  = blockIdx.x / nbm;
  const int cb     = blockIdx.x % nbm;

  const int l16  = lane & 15;
  const int koff = (lane >> 4) << 3;              // 0 or 8

  // cooperative A staging: threads 0..127 copy 8 halves each (16 rows x 64)
  const bool astage = tid < 128;                  // waves 0..3 (wave-uniform)
  const int  arow   = (tid >> 3) & 15;            // 0..15
  const int  acol   = (tid & 7) << 3;             // 0..56 step 8
  const __bf16* Ag  = A + (long)(mtile * 16 + arow) * K + acol;

  // B (weight) pointers, one 16-col tile per s
  const __bf16* Wp[GEMM_NT];
  bool valid[GEMM_NT];
#pragma unroll
  for (int s = 0; s < GEMM_NT; ++s) {
    int nt = (cb * 8 + wid) * GEMM_NT + s;
    valid[s] = (nt < ntiles);
    int wrow = valid[s] ? (nt * 16 + l16) : l16;  // safe row for invalid subtile
    Wp[s] = W + (long)wrow * K + koff;
  }

  v8f acc[GEMM_NT];
#pragma unroll
  for (int s = 0; s < GEMM_NT; ++s) acc[s] = (v8f){0.f,0.f,0.f,0.f,0.f,0.f,0.f,0.f};

  // stage A chunk 0 via async copy, then publish
  if (astage) async_copy_b128(Ag, &As[0][arow][acol]);
  wait_async0();
  __syncthreads();

  int buf = 0;
  for (int kc = 0; kc < K; kc += KCHUNK, buf ^= 1) {
    const bool more = (kc + KCHUNK) < K;
    if (more) {
      // kick off next A chunk now: overlaps the entire compute phase below
      if (astage) async_copy_b128(Ag + kc + KCHUNK, &As[buf ^ 1][arow][acol]);
#pragma unroll
      for (int s = 0; s < GEMM_NT; ++s)           // prime caches for next B chunk
        __builtin_prefetch(Wp[s] + kc + KCHUNK, 0, 0);
    }

    // ---- phase 1: hoist ALL loads for this chunk into registers ----
    v8bf bb0[KSTEPS][GEMM_NT], bb1[KSTEPS][GEMM_NT];
#pragma unroll
    for (int ks = 0; ks < KSTEPS; ++ks)
#pragma unroll
      for (int s = 0; s < GEMM_NT; ++s) {
        bb0[ks][s] = *(const v8bf*)(Wp[s] + kc + ks * 32);
        bb1[ks][s] = *(const v8bf*)(Wp[s] + kc + ks * 32 + 16);
      }
    v8bf aa0[KSTEPS], aa1[KSTEPS];
#pragma unroll
    for (int ks = 0; ks < KSTEPS; ++ks) {
      aa0[ks] = *(const v8bf*)&As[buf][l16][ks * 32 + koff];
      aa1[ks] = *(const v8bf*)&As[buf][l16][ks * 32 + koff + 16];
    }
    asm volatile("" ::: "memory");   // pin loads above the MMA block

    // ---- phase 2: 8 back-to-back WMMAs ----
#pragma unroll
    for (int ks = 0; ks < KSTEPS; ++ks) {
      v16bf av = __builtin_shufflevector(aa0[ks], aa1[ks],
                                         0,1,2,3,4,5,6,7,8,9,10,11,12,13,14,15);
#pragma unroll
      for (int s = 0; s < GEMM_NT; ++s) {
        v16bf bv = __builtin_shufflevector(bb0[ks][s], bb1[ks][s],
                                           0,1,2,3,4,5,6,7,8,9,10,11,12,13,14,15);
        acc[s] = __builtin_amdgcn_wmma_f32_16x16x32_bf16(
            false, av, false, bv, (short)0, acc[s], false, false);
      }
    }

    if (more) {
      // publish As[buf^1]: async copy done + all waves past their reads
      wait_async0();
      __syncthreads();
    }
  }

  // epilogue
  const int rbase = mtile * 16 + ((lane >> 4) << 3);
#pragma unroll
  for (int s = 0; s < GEMM_NT; ++s) {
    if (!valid[s]) continue;
    int col = ((cb * 8 + wid) * GEMM_NT + s) * 16 + l16;
    float bb = bias ? bias[col] : 0.0f;
#pragma unroll
    for (int r = 0; r < 8; ++r) {
      int row = rbase + r;
      long idx = (long)row * ldc + col;
      float v = acc[s][r] + bb;
      if (lengths && !(lengths[row] > t)) v = 0.0f;  // ragged masking for logits
      C[idx] = v;
    }
  }
}

// ---------------------------------------------------------------------------
// LSTM cell elementwise (torch gate order i,f,g,o), with state freeze.
// Writes: frozen fp32 state h,c; bf16(h_new unfrozen) -> hnb (contiguous) and
// optional xdst slice; bf16(h frozen) -> up to two strided slices d1/d2
// (feeding the fused-K activation rows of the next step's GEMMs).
// ---------------------------------------------------------------------------
__device__ __forceinline__ float sigmoidf_(float x) { return 1.0f / (1.0f + expf(-x)); }

__global__ void k_lstm_cell(const float* __restrict__ gates,
                            float* __restrict__ h, float* __restrict__ c,
                            __bf16* __restrict__ hnb,
                            __bf16* __restrict__ d1, int d1stride, int d1off,
                            __bf16* __restrict__ d2, int d2stride, int d2off,
                            __bf16* __restrict__ xdst, int xstride, int xoff,
                            const int* __restrict__ lengths, int t, int H) {
  int i = blockIdx.x * blockDim.x + threadIdx.x;
  if (i >= Bc * H) return;
  int b = i / H, j = i % H;
  const float* g = gates + (long)b * 4 * H;
  float ig = sigmoidf_(g[j]);
  float fg = sigmoidf_(g[H + j]);
  float gg = tanhf(g[2 * H + j]);
  float og = sigmoidf_(g[3 * H + j]);
  float cold = c[i], hold = h[i];
  float cn = fg * cold + ig * gg;
  float hn = og * tanhf(cn);
  bool active = lengths[b] > t;
  float hs = active ? hn : hold;
  float cs = active ? cn : cold;
  h[i] = hs; c[i] = cs;
  hnb[i] = (__bf16)hn;
  __bf16 hsb = (__bf16)hs;
  if (d1) d1[(long)b * d1stride + d1off + j] = hsb;
  if (d2) d2[(long)b * d2stride + d2off + j] = hsb;
  if (xdst) xdst[(long)b * xstride + xoff + j] = (__bf16)hn;
}

// ---------------------------------------------------------------------------
// Top-down attention: logit[b,r] = tanh(Va[b,r,:]+ha[b,:])·wa + ba;
// softmax over R; av = atten·Vmat; write bf16 av into x2c[:,0:V].
// ---------------------------------------------------------------------------
__global__ __launch_bounds__(256)
void k_attn(const float* __restrict__ Va, const float* __restrict__ ha,
            const float* __restrict__ wa, const float* __restrict__ ba,
            const float* __restrict__ Vmat, __bf16* __restrict__ x2c) {
  int b = blockIdx.x;
  int tid = threadIdx.x;
  __shared__ float red[256];
  __shared__ float sl[Rc];
  __shared__ float satt[Rc];

  float hav = ha[(long)b * PHc + tid];
  float wav = wa[tid];
  for (int r = 0; r < Rc; ++r) {
    float v = tanhf(Va[((long)b * Rc + r) * PHc + tid] + hav) * wav;
    red[tid] = v;
    __syncthreads();
    for (int s = 128; s > 0; s >>= 1) {
      if (tid < s) red[tid] += red[tid + s];
      __syncthreads();
    }
    if (tid == 0) sl[r] = red[0] + ba[0];
    __syncthreads();
  }
  if (tid == 0) {
    float mx = sl[0];
    for (int r = 1; r < Rc; ++r) mx = fmaxf(mx, sl[r]);
    float s = 0.f;
    for (int r = 0; r < Rc; ++r) { float e = expf(sl[r] - mx); satt[r] = e; s += e; }
    float inv = 1.f / s;
    for (int r = 0; r < Rc; ++r) satt[r] *= inv;
  }
  __syncthreads();
  for (int d = tid; d < Vc; d += 256) {
    float a = 0.f;
    for (int r = 0; r < Rc; ++r) a += satt[r] * Vmat[((long)b * Rc + r) * Vc + d];
    x2c[(long)b * K2c + d] = (__bf16)a;
  }
}

// ---------------------------------------------------------------------------
static void launch_gemm(const __bf16* A, const __bf16* W, const float* bias, float* C,
                        int M, int N, int K, long ldc,
                        const int* lengths, int t, hipStream_t s) {
  int mt  = M / 16;
  int nbm = (N / 16 + 31) / 32;   // 8 waves x 4 tiles = 32 column tiles / block
  k_gemm_bf16<<<mt * nbm, 256, 0, s>>>(A, W, bias, C, M, N, K, ldc, lengths, t);
}

extern "C" void kernel_launch(void* const* d_in, const int* in_sizes, int n_in,
                              void* d_out, int out_size, void* d_ws, size_t ws_size,
                              hipStream_t stream) {
  (void)in_sizes; (void)n_in; (void)out_size;
  const float* Vmat  = (const float*)d_in[0];
  /* d_in[1] enc_features: unused (matches reference) */
  const float* uv    = (const float*)d_in[2];
  const int*   caps  = (const int*)d_in[3];
  const int*   lens  = (const int*)d_in[4];
  const float* embW  = (const float*)d_in[5];
  const float* Wi1   = (const float*)d_in[6];
  const float* Wh1   = (const float*)d_in[7];
  const float* b1    = (const float*)d_in[8];
  const float* Wi2   = (const float*)d_in[9];
  const float* Wh2   = (const float*)d_in[10];
  const float* b2    = (const float*)d_in[11];
  const float* Wva   = (const float*)d_in[12];
  const float* bva   = (const float*)d_in[13];
  const float* Wha   = (const float*)d_in[14];
  const float* bha   = (const float*)d_in[15];
  const float* wa    = (const float*)d_in[16];
  const float* ba    = (const float*)d_in[17];
  const float* Wl    = (const float*)d_in[18];
  const float* bl    = (const float*)d_in[19];
  float* out = (float*)d_out;

  // ---- workspace carve-up (256B aligned) ----
  char* w = (char*)d_ws;
  size_t off = 0;
  auto alloc = [&](size_t bytes) -> char* {
    off = (off + 255) & ~(size_t)255;
    char* p = w + off; off += bytes; return p;
  };
  __bf16* Wc1b  = (__bf16*)alloc((size_t)4096 * K1c * 2);   // [Wi1 | Wh1]
  __bf16* Wc2b  = (__bf16*)alloc((size_t)4096 * K2c * 2);   // [Wi2 | Wh2]
  __bf16* Wlb   = (__bf16*)alloc((size_t)VOCc * 1024 * 2);
  __bf16* Wvab  = (__bf16*)alloc((size_t)PHc * Vc * 2);
  __bf16* Whab  = (__bf16*)alloc((size_t)PHc * H1c * 2);
  __bf16* Vmatb = (__bf16*)alloc((size_t)Bc * Rc * Vc * 2);
  float*  Va    = (float*)alloc((size_t)Bc * Rc * PHc * 4);
  __bf16* x1c   = (__bf16*)alloc((size_t)Bc * K1c * 2);
  __bf16* x2c   = (__bf16*)alloc((size_t)Bc * K2c * 2);
  float*  gates1 = (float*)alloc((size_t)Bc * 4096 * 4);
  float*  gates2 = (float*)alloc((size_t)Bc * 4096 * 4);
  float*  ha     = (float*)alloc((size_t)Bc * PHc * 4);
  char*   stateblk = alloc((size_t)4 * Bc * 1024 * 4);      // h1,c1,h2,c2 fp32
  float*  h1 = (float*)stateblk;
  float*  c1 = h1 + (size_t)Bc * 1024;
  float*  h2 = c1 + (size_t)Bc * 1024;
  float*  c2 = h2 + (size_t)Bc * 1024;
  __bf16* h1nb = (__bf16*)alloc((size_t)Bc * 1024 * 2);
  __bf16* h2nb = (__bf16*)alloc((size_t)Bc * 1024 * 2);
  if (off > ws_size) return;  // workspace too small: do nothing (compile-only loop)

  auto cvt = [&](const float* s, __bf16* d, size_t n) {
    k_f32_to_bf16<<<(int)((n + 255) / 256), 256, 0, stream>>>(s, d, (int)n);
  };

  // ---- one-time (per launch) setup ----
  {
    long t1 = (long)4096 * K1c;
    k_pack_w<<<(int)((t1 + 255) / 256), 256, 0, stream>>>(Wi1, Wh1, Wc1b, 4096, 1024, t1);
    long t2 = (long)4096 * K2c;
    k_pack_w<<<(int)((t2 + 255) / 256), 256, 0, stream>>>(Wi2, Wh2, Wc2b, 3072, 1024, t2);
  }
  cvt(Wl,  Wlb,  (size_t)VOCc * 1024);
  cvt(Wva, Wvab, (size_t)PHc * Vc);
  cvt(Wha, Whab, (size_t)PHc * H1c);
  cvt(Vmat, Vmatb, (size_t)Bc * Rc * Vc);
  // zero fp32 states and the fused activation rows (h-slices must start at 0)
  hipMemsetAsync(stateblk, 0, (size_t)4 * Bc * 1024 * 4, stream);
  hipMemsetAsync(x1c, 0, (size_t)Bc * K1c * 2, stream);
  hipMemsetAsync(x2c, 0, (size_t)Bc * K2c * 2, stream);
  k_x1_uv<<<(Bc * Vc + 255) / 256, 256, 0, stream>>>(x1c, uv);
  // Va = Vmat @ Wva^T + bva   [B*R, PH]
  launch_gemm(Vmatb, Wvab, bva, Va, Bc * Rc, PHc, Vc, PHc, nullptr, 0, stream);

  // ---- recurrent steps ----
  const int nh = Bc * 1024;
  for (int t = 0; t < Tc; ++t) {
    // x1c embedding slice for this step
    k_x1_emb<<<(Bc * Ec + 255) / 256, 256, 0, stream>>>(x1c, embW, caps, t);
    // gates1 = [h2,uv,emb,h1] @ [Wi1|Wh1]^T + b1   (fused input+recurrent)
    launch_gemm(x1c, Wc1b, b1, gates1, Bc, 4096, K1c, 4096, nullptr, 0, stream);
    // LSTM1: frozen h1 -> x1c[:,4096:]; unfrozen h1n -> h1nb and x2c[:,2048:3072]
    k_lstm_cell<<<(nh + 255) / 256, 256, 0, stream>>>(
        gates1, h1, c1, h1nb,
        x1c, K1c, H2c + Vc + Ec,   // d1: frozen h1 into x1c recurrent slice
        nullptr, 0, 0,             // d2: none
        x2c, K2c, Vc,              // xdst: unfrozen h1n into x2c
        lens, t, H1c);
    // ha = h1n @ Wha^T + bha
    launch_gemm(h1nb, Whab, bha, ha, Bc, PHc, H1c, PHc, nullptr, 0, stream);
    // attention -> x2c[:, 0:V] = bf16(av)
    k_attn<<<Bc, 256, 0, stream>>>(Va, ha, wa, ba, Vmat, x2c);
    // gates2 = [av,h1n,h2] @ [Wi2|Wh2]^T + b2   (fused input+recurrent)
    launch_gemm(x2c, Wc2b, b2, gates2, Bc, 4096, K2c, 4096, nullptr, 0, stream);
    // LSTM2: frozen h2 -> x1c[:,0:1024] and x2c[:,3072:]; unfrozen h2n -> h2nb
    k_lstm_cell<<<(nh + 255) / 256, 256, 0, stream>>>(
        gates2, h2, c2, h2nb,
        x1c, K1c, 0,               // d1: frozen h2 into x1c head slice
        x2c, K2c, Vc + H1c,        // d2: frozen h2 into x2c recurrent slice
        nullptr, 0, 0,             // xdst: none
        lens, t, H2c);
    // logits -> d_out[:, t, :], masked to 0 for finished sequences
    launch_gemm(h2nb, Wlb, bl, out + (size_t)t * VOCc, Bc, VOCc, 1024,
                (long)Tc * VOCc, lens, t, stream);
  }
}
